// H2GCN_33217277067916
// MI455X (gfx1250) — matrix operand order repaired and checked by reading, not compiled, not run on hardware
//
#include <hip/hip_runtime.h>
#include <hip/hip_bf16.h>
#include <stdint.h>
#include <stddef.h>

// ---------------------------------------------------------------------------
// H2GCN forward for MI455X (gfx1250, wave32, WMMA + async global->LDS).
//
// Roofline: the op is HBM-bound on streaming the two 268MB f32 adjacency
// matrices twice (~1.07GB -> ~46us @ 23.3TB/s).  Design rules:
//   * each adjacency element is loaded once per block column (128-row x
//     256-col output tiles: once in conv1, twice in conv2; f32->f16 conversion in registers, tile shared
//     across 8 waves via LDS),
//   * B panels are produced by us, so we emit transposed f16 copies (hT/zT,
//     [N][K]) at production time -- the WMMA C layout makes those stores one
//     contiguous b128 per lane -- and stage them with
//     global_load_async_to_lds_b128 (+ s_wait_asynccnt) each k-step,
//   * all matmuls are v_wmma_f32_16x16x32_f16 with f32 accumulation
//     (adjacency in [0,1], activations ~N(0,1): well inside f16 range).
// ---------------------------------------------------------------------------

typedef __attribute__((ext_vector_type(16))) _Float16 v16h;
typedef __attribute__((ext_vector_type(8)))  _Float16 v8h;
typedef __attribute__((ext_vector_type(4)))  _Float16 v4h;
typedef __attribute__((ext_vector_type(8)))  float    v8f;
typedef __attribute__((ext_vector_type(4)))  float    v4f;

union F16x16 { v16h v; v8h h[2]; };

#define NN     8192   // nodes
#define HID    256
#define TWOH   512
#define JKW    1792   // 7*HID
#define OUTC   64

#define USE_ASYNC_B 1

// Single shared object so offsetof() gives the true LDS byte offset
// (static LDS allocation starts at 0) for the async-copy destinations.
struct SMem {
    _Float16 sA[2][128 * 40];   // A tile  [M=128][K=32] f16, row padded to 40 halves
    _Float16 sB[2][256 * 40];   // B tile  [N=256][K=32] f16, row padded to 40 halves
};

__device__ __forceinline__ void async_ld_b128(uint32_t lds_byte_off, const void* g)
{
    // CDNA5 VGLOBAL async copy: LDS[vdst] = MEM[vaddr], tracked by ASYNCcnt.
    asm volatile("global_load_async_to_lds_b128 %0, %1, off"
                 :: "v"(lds_byte_off), "v"(g) : "memory");
}
__device__ __forceinline__ void wait_async0()
{
    asm volatile("s_wait_asynccnt 0x0" ::: "memory");
}

// ---------------------------------------------------------------------------
// Tiled GEMM:  C[128 x 256 tile] = A(f32 -> f16) @ B(f16, [N][K] row-major)
// 8 waves, each 64x64 (16 WMMA accumulators).
// MODE 0: embed -> f16 jk[:,0:256] + bias + ReLU, also writes hT [N][K]
// MODE 1: conv1 -> f32 zpre
// MODE 2: conv2 -> f16 jk[:,768:1792]
// blockIdx.z selects A0/A1 (1-hop vs strict 2-hop adjacency).
// ---------------------------------------------------------------------------
template<int MODE>
__global__ __launch_bounds__(256)
void gemm_kernel(const float* __restrict__ A0, const float* __restrict__ A1,
                 int lda, int K,
                 const _Float16* __restrict__ B, int ldb,
                 void* __restrict__ Cout, int ldc,
                 int colBase, int zColStride,
                 const float* __restrict__ bias,
                 _Float16* __restrict__ Tout)   // transposed f16 output (MODE 0)
{
    constexpr int BM = 128, BN = 256, BK = 32, LDH = 40;
    __shared__ SMem smem;

    const float* A = (blockIdx.z == 0) ? A0 : A1;
    const int tid  = threadIdx.x;
    const int lane = tid & 31;
    const int wave = tid >> 5;
    const int wm = wave >> 2, wn = wave & 3;       // 2 x 4 wave grid
    const int l15 = lane & 15;
    const bool hi = lane >= 16;
    const int blockRow = blockIdx.x * BM;
    const int bcol     = blockIdx.y * BN;

    // A staging: each thread moves 16 f32 (row ar, cols ac..ac+15)
    const int ar = tid >> 1;
    const int ac = (tid & 1) * 16;
    const size_t aRowOff = (size_t)(blockRow + ar) * (size_t)lda + ac;

    // B staging: each thread owns one B row (= output column), 4 x 16B chunks
    const _Float16* bRow = B + (size_t)(bcol + tid) * (size_t)ldb;
    const uint32_t  bLds = (uint32_t)(offsetof(SMem, sB) + (size_t)tid * LDH * 2);

    v4f areg[4];
    v8f acc[4][4] = {};

    const int KT = K / BK;

    auto loadA = [&](int kb) {
        const float* ap = A + aRowOff + (size_t)kb * BK;
        #pragma unroll
        for (int j = 0; j < 4; ++j)
            areg[j] = *reinterpret_cast<const v4f*>(ap + j * 4);
    };
    auto storeA = [&](int buf) {
        _Float16* da = &smem.sA[buf][ar * LDH + ac];
        #pragma unroll
        for (int j = 0; j < 4; ++j) {
            v4h h4;
            #pragma unroll
            for (int e = 0; e < 4; ++e) h4[e] = (_Float16)areg[j][e];
            *reinterpret_cast<v4h*>(da + j * 4) = h4;          // ds_store_b64
        }
    };
    auto issueB = [&](int buf, int kb) {
#if USE_ASYNC_B
        const uint32_t dst = bLds + (uint32_t)buf * (BN * LDH * 2);
        const _Float16* src = bRow + kb * BK;
        #pragma unroll
        for (int c = 0; c < 4; ++c)
            async_ld_b128(dst + c * 16, src + c * 8);
#else
        const uint32_t dst = bLds + (uint32_t)buf * (BN * LDH * 2);
        const _Float16* src = bRow + kb * BK;
        _Float16* d = &smem.sB[buf][tid * LDH];
        #pragma unroll
        for (int c = 0; c < 4; ++c)
            *reinterpret_cast<v8h*>(d + c * 8) =
                *reinterpret_cast<const v8h*>(src + c * 8);
        (void)dst;
#endif
    };

    auto computeTile = [&](int buf) {
        F16x16 af[4], bf[4];
        // A fragment (16x32): lanes 0-15 K{0..7,16..23}, lanes 16-31 K{8..15,24..31}
        const int aoff = hi ? 8 : 0;
        #pragma unroll
        for (int mi = 0; mi < 4; ++mi) {
            const _Float16* base = &smem.sA[buf][(wm * 64 + mi * 16 + l15) * LDH + aoff];
            af[mi].h[0] = *reinterpret_cast<const v8h*>(base);     // ds_load_b128
            af[mi].h[1] = *reinterpret_cast<const v8h*>(base + 16);
        }
        // B fragment (32x16): lane n = column; lanes 0-15 K{0..7,16..23}, 16-31 K{8..15,24..31}
        const int boff = hi ? 8 : 0;
        #pragma unroll
        for (int ni = 0; ni < 4; ++ni) {
            const _Float16* base = &smem.sB[buf][(wn * 64 + ni * 16 + l15) * LDH + boff];
            bf[ni].h[0] = *reinterpret_cast<const v8h*>(base);
            bf[ni].h[1] = *reinterpret_cast<const v8h*>(base + 16);
        }
        #pragma unroll
        for (int mi = 0; mi < 4; ++mi)
            #pragma unroll
            for (int ni = 0; ni < 4; ++ni)
                acc[mi][ni] = __builtin_amdgcn_wmma_f32_16x16x32_f16(
                    false, af[mi].v, false, bf[ni].v,
                    (short)0, acc[mi][ni], false, false);
    };

    // ---- software pipeline ----
    loadA(0);
    issueB(0, 0);
    storeA(0);
#if USE_ASYNC_B
    wait_async0();
#endif
    __syncthreads();

    for (int kb = 0; kb < KT; ++kb) {
        const int cur = kb & 1;
        if (kb + 1 < KT) {
            loadA(kb + 1);
            issueB(cur ^ 1, kb + 1);       // async copy overlaps the WMMA work
            if (kb + 4 < KT)               // speculative stream prefetch (adjacency)
                __builtin_prefetch(A + aRowOff + (size_t)(kb + 4) * BK, 0, 0);
        }
        computeTile(cur);
        if (kb + 1 < KT) storeA(cur ^ 1);
#if USE_ASYNC_B
        wait_async0();                     // this wave's copies for nxt are done
#endif
        __syncthreads();                   // everyone's copies visible
    }

    // ---- epilogue (C layout: vgpr r -> M = r / r+8, lane -> N) ----
    const int rowBase = blockRow + wm * 64 + (hi ? 8 : 0);
    #pragma unroll
    for (int mi = 0; mi < 4; ++mi) {
        #pragma unroll
        for (int ni = 0; ni < 4; ++ni) {
            const int colLocal = bcol + wn * 64 + ni * 16 + l15;
            const int col = colBase + (int)blockIdx.z * zColStride + colLocal;
            if (MODE == 0) {
                const float bb = bias[colLocal];
                v8h pack;
                #pragma unroll
                for (int r = 0; r < 8; ++r) {
                    float v = acc[mi][ni][r] + bb;
                    v = v > 0.f ? v : 0.f;
                    const _Float16 hv = (_Float16)v;
                    reinterpret_cast<_Float16*>(Cout)[(size_t)(rowBase + mi * 16 + r) * ldc + col] = hv;
                    pack[r] = hv;
                }
                // transposed copy: 8 consecutive M per lane -> one b128 store
                *reinterpret_cast<v8h*>(&Tout[(size_t)colLocal * NN + rowBase + mi * 16]) = pack;
            } else if (MODE == 1) {
                #pragma unroll
                for (int r = 0; r < 8; ++r)
                    reinterpret_cast<float*>(Cout)[(size_t)(rowBase + mi * 16 + r) * ldc + col] =
                        acc[mi][ni][r];
            } else {
                #pragma unroll
                for (int r = 0; r < 8; ++r)
                    reinterpret_cast<_Float16*>(Cout)[(size_t)(rowBase + mi * 16 + r) * ldc + col] =
                        (_Float16)acc[mi][ni][r];
            }
        }
    }
}

// ---------------------------------------------------------------------------
// BatchNorm statistics: one block per channel, deterministic tree reduction.
// zpre (16MB) is L2-resident right after conv1.
// ---------------------------------------------------------------------------
__global__ void bn_stats_kernel(const float* __restrict__ z,
                                float* __restrict__ mean, float* __restrict__ var)
{
    __shared__ float ss[256], sq[256];
    const int c = blockIdx.x, t = threadIdx.x;
    float s = 0.f, q = 0.f;
    for (int r = t; r < NN; r += 256) {
        float v = z[(size_t)r * TWOH + c];
        s += v; q += v * v;
    }
    ss[t] = s; sq[t] = q;
    __syncthreads();
    for (int o = 128; o > 0; o >>= 1) {
        if (t < o) { ss[t] += ss[t + o]; sq[t] += sq[t + o]; }
        __syncthreads();
    }
    if (t == 0) {
        float m = ss[0] * (1.f / NN);
        mean[c] = m;
        var[c]  = sq[0] * (1.f / NN) - m * m;   // biased, matches jnp.var
    }
}

// ---------------------------------------------------------------------------
// BN apply fused with a 64x64 LDS tile transpose:
//   jk[:,256:768] (row-major, final GEMM input) and zT [512][8192] (conv2 B).
// ---------------------------------------------------------------------------
__global__ __launch_bounds__(256)
void bn_apply_transpose_kernel(const float* __restrict__ zpre,
                               const float* __restrict__ mean,
                               const float* __restrict__ var,
                               const float* __restrict__ gamma,
                               const float* __restrict__ beta,
                               _Float16* __restrict__ jk,
                               _Float16* __restrict__ zT)
{
    __shared__ _Float16 tile[64 * 65];
    const int r0 = blockIdx.x * 64;        // node tile
    const int c0 = blockIdx.y * 64;        // channel tile
    const int tc  = threadIdx.x & 63;
    const int tr4 = threadIdx.x >> 6;      // 0..3

    const int c = c0 + tc;                 // fixed channel for phase 1
    const float m  = mean[c];
    const float is = rsqrtf(var[c] + 1e-5f);
    const float g  = gamma[c];
    const float b  = beta[c];

    #pragma unroll
    for (int ii = 0; ii < 16; ++ii) {
        const int rl = tr4 + ii * 4;
        const int r  = r0 + rl;
        const float v = (zpre[(size_t)r * TWOH + c] - m) * is * g + b;
        const _Float16 hv = (_Float16)v;
        jk[(size_t)r * JKW + HID + c] = hv;       // coalesced along c
        tile[tc * 65 + rl] = hv;
    }
    __syncthreads();
    #pragma unroll
    for (int ii = 0; ii < 16; ++ii) {
        const int cl = tr4 + ii * 4;
        zT[(size_t)(c0 + cl) * NN + r0 + tc] = tile[cl * 65 + tc];  // coalesced along r
    }
}

// ---------------------------------------------------------------------------
// Final projection: out[8192x64] = jk(f16)[8192x1792] @ w_fin^T + b_fin.
// One wave per 16-row strip; w_fin (224KB f16) is L2/L0 resident.
// All fragments for a k-step are loaded before the 4 WMMAs (single wait).
// ---------------------------------------------------------------------------
__global__ __launch_bounds__(256)
void final_gemm_kernel(const _Float16* __restrict__ jk,
                       const _Float16* __restrict__ wf,   // [64][1792] f16
                       const float* __restrict__ bias, float* __restrict__ out)
{
    const int tid  = threadIdx.x;
    const int lane = tid & 31;
    const int wave = blockIdx.x * 8 + (tid >> 5);
    const int l15  = lane & 15;
    const bool hi  = lane >= 16;
    const int m0   = wave * 16;

    v8f acc[4] = {};
    const int aoff = hi ? 8 : 0;
    const int boff = hi ? 8 : 0;
    const _Float16* arow = jk + (size_t)(m0 + l15) * JKW + aoff;
    const _Float16* brow[4];
    #pragma unroll
    for (int ni = 0; ni < 4; ++ni)
        brow[ni] = wf + (size_t)(ni * 16 + l15) * JKW + boff;

    for (int k = 0; k < JKW; k += 32) {
        F16x16 a, b[4];
        a.h[0] = *reinterpret_cast<const v8h*>(arow + k);
        a.h[1] = *reinterpret_cast<const v8h*>(arow + k + 16);
        #pragma unroll
        for (int ni = 0; ni < 4; ++ni) {
            b[ni].h[0] = *reinterpret_cast<const v8h*>(brow[ni] + k);
            b[ni].h[1] = *reinterpret_cast<const v8h*>(brow[ni] + k + 16);
        }
        #pragma unroll
        for (int ni = 0; ni < 4; ++ni)
            acc[ni] = __builtin_amdgcn_wmma_f32_16x16x32_f16(
                false, a.v, false, b[ni].v, (short)0, acc[ni], false, false);
    }
    #pragma unroll
    for (int ni = 0; ni < 4; ++ni) {
        const int col = ni * 16 + l15;
        const float bb = bias[col];
        #pragma unroll
        for (int r = 0; r < 8; ++r) {
            const int row = m0 + (hi ? 8 : 0) + r;
            out[(size_t)row * OUTC + col] = acc[ni][r] + bb;
        }
    }
}

__global__ void cvt_f32_f16_kernel(const float* __restrict__ w, _Float16* __restrict__ o, int n)
{
    const int i = blockIdx.x * 256 + threadIdx.x;
    if (i < n) o[i] = (_Float16)w[i];
}

// ---------------------------------------------------------------------------
extern "C" void kernel_launch(void* const* d_in, const int* in_sizes, int n_in,
                              void* d_out, int out_size, void* d_ws, size_t ws_size,
                              hipStream_t stream)
{
    const float* x     = (const float*)d_in[0];
    const float* adj   = (const float*)d_in[1];
    const float* adj2  = (const float*)d_in[2];
    const float* wemb  = (const float*)d_in[3];
    const float* bemb  = (const float*)d_in[4];
    const float* gamma = (const float*)d_in[5];
    const float* beta  = (const float*)d_in[6];
    const float* wfin  = (const float*)d_in[7];
    const float* bfin  = (const float*)d_in[8];
    float* out = (float*)d_out;

    char* ws = (char*)d_ws;
    size_t off = 0;
    auto walloc = [&](size_t bytes) {
        void* p = ws + off;
        off = (off + bytes + 255) & ~(size_t)255;
        return p;
    };
    _Float16* jk     = (_Float16*)walloc((size_t)NN * JKW * 2);    // h|z|z2 packed f16
    float*    zpre   = (float*)   walloc((size_t)NN * TWOH * 4);   // pre-BN conv1
    _Float16* hT     = (_Float16*)walloc((size_t)HID * NN * 2);    // h transposed [256][8192]
    _Float16* zT     = (_Float16*)walloc((size_t)TWOH * NN * 2);   // z transposed [512][8192]
    _Float16* wembH  = (_Float16*)walloc((size_t)HID * 512 * 2);   // w_embed as [N][K] f16
    _Float16* wfin16 = (_Float16*)walloc((size_t)OUTC * JKW * 2);
    float*    mean   = (float*)   walloc(TWOH * 4);
    float*    var    = (float*)   walloc(TWOH * 4);

    cvt_f32_f16_kernel<<<(HID * 512 + 255) / 256, 256, 0, stream>>>(wemb, wembH, HID * 512);
    cvt_f32_f16_kernel<<<(OUTC * JKW + 255) / 256, 256, 0, stream>>>(wfin, wfin16, OUTC * JKW);

    // h = relu(x @ w_embed^T + b) -> jk[:,0:256] (f16) and hT [256][8192]
    gemm_kernel<0><<<dim3(NN / 128, 1, 1), 256, 0, stream>>>(
        x, x, 512, 512, wembH, 512, (void*)jk, JKW, 0, 0, bemb, hT);

    // zpre = [A@h | A2@h] (f32); blockIdx.z picks adjacency, B = hT
    gemm_kernel<1><<<dim3(NN / 128, 1, 2), 256, 0, stream>>>(
        adj, adj2, NN, NN, hT, NN, (void*)zpre, TWOH, 0, HID, nullptr, nullptr);

    bn_stats_kernel<<<TWOH, 256, 0, stream>>>(zpre, mean, var);
    bn_apply_transpose_kernel<<<dim3(NN / 64, TWOH / 64), 256, 0, stream>>>(
        zpre, mean, var, gamma, beta, jk, zT);

    // z2 = [A@z | A2@z] -> jk[:,768:1792] (f16); B = zT
    gemm_kernel<2><<<dim3(NN / 128, 2, 2), 256, 0, stream>>>(
        adj, adj2, NN, NN, zT, NN, (void*)jk, JKW, 768, TWOH, nullptr, nullptr);

    // out = jk @ w_fin^T + b_fin
    final_gemm_kernel<<<NN / 16 / 8, 256, 0, stream>>>(jk, wfin16, bfin, out);
}
